// GPT2Attention_49709951483946
// MI455X (gfx1250) — matrix operand order repaired and checked by reading, not compiled, not run on hardware
//
#include <hip/hip_runtime.h>
#include <cstdint>
#include <cstddef>

typedef unsigned short u16;
typedef __attribute__((ext_vector_type(16))) __bf16 v16bf;
typedef __attribute__((ext_vector_type(8)))  float  v8f;
typedef __attribute__((ext_vector_type(4)))  int    v4i;

#define DEV __device__ __forceinline__

constexpr int BATCH = 2;
constexpr int SEQ   = 2048;
constexpr int EMB   = 2048;
constexpr int NH    = 16;
constexpr int NKV   = 8;
constexpr int DH    = 128;
constexpr int MTOK  = BATCH * SEQ;          // 4096 tokens
constexpr int NQKV  = EMB + 2 * NKV * DH;   // 4096 (q:2048, k:1024, v:1024)
constexpr float ATT_SCALE = 0.08838834764831845f; // 1/sqrt(128)
constexpr float QKGAIN = 1.15f;
constexpr float RMSEPS = 1e-6f;

// ---- CDNA5 async global->LDS support (guarded; falls back cleanly) --------
#if defined(__has_builtin)
#  if __has_builtin(__builtin_amdgcn_global_load_async_to_lds_b128)
#    define HAVE_ASYNC_LDS 1
#  endif
#endif
#ifndef HAVE_ASYNC_LDS
#  define HAVE_ASYNC_LDS 0
#endif

#if HAVE_ASYNC_LDS
typedef __attribute__((address_space(1))) v4i gas_v4i;  // global
typedef __attribute__((address_space(3))) v4i las_v4i;  // LDS

DEV void async_copy16(const u16* g, u16* l) {
  // GLOBAL_LOAD_ASYNC_TO_LDS_B128: 16B/lane global -> LDS, tracked by ASYNCcnt
  __builtin_amdgcn_global_load_async_to_lds_b128(
      (gas_v4i*)(v4i*)(void*)g, (las_v4i*)(v4i*)(void*)l, 0, 0);
}
DEV void wait_async_le6() {
#  if __has_builtin(__builtin_amdgcn_s_wait_asynccnt)
  __builtin_amdgcn_s_wait_asynccnt(6);
#  else
  asm volatile("s_wait_asynccnt 0x6" ::: "memory");
#  endif
}
DEV void wait_async_le3() {
#  if __has_builtin(__builtin_amdgcn_s_wait_asynccnt)
  __builtin_amdgcn_s_wait_asynccnt(3);
#  else
  asm volatile("s_wait_asynccnt 0x3" ::: "memory");
#  endif
}
DEV void wait_async_0() {
#  if __has_builtin(__builtin_amdgcn_s_wait_asynccnt)
  __builtin_amdgcn_s_wait_asynccnt(0);
#  else
  asm volatile("s_wait_asynccnt 0x0" ::: "memory");
#  endif
}
#endif

union Frag {
  uint4 u[2];
  v16bf v;
};

DEV u16 f2bf(float f) {
  union { float f; unsigned u; } c; c.f = f;
  unsigned u = c.u;
  unsigned r = (u + 0x7fffu + ((u >> 16) & 1u)) >> 16; // RNE
  return (u16)r;
}

DEV v8f wmma_bf16(const Frag& a, const Frag& b, v8f c) {
  // D = A(16x32 bf16) * B(32x16 bf16) + C(16x16 f32)
  return __builtin_amdgcn_wmma_f32_16x16x32_bf16(false, a.v, false, b.v,
                                                 (short)0, c, false, false);
}

// --------------------------- f32 -> bf16 cast ------------------------------
__global__ void cast_bf16_kernel(const float* __restrict__ src,
                                 u16* __restrict__ dst, int n) {
  int i = blockIdx.x * blockDim.x + threadIdx.x;
  if (i < n) dst[i] = f2bf(src[i]);
}

// ----------------- GEMM: C[M,N] = A[M,K] * B[N,K]^T (bf16->f32) ------------
// Block tile 64(M) x 128(N), BK=32, 256 threads = 8 waves (2x4), wave=32x32.
// Triple-buffered LDS with async global->LDS; 2 tiles in flight.
DEV void mma_step(const u16* as, const u16* bs, int wm, int wn, int lane16,
                  int half, v8f acc[2][2]) {
  Frag a[2], b;
#pragma unroll
  for (int mt = 0; mt < 2; mt++) {  // A frag: lane=M row, K runs 8h / 16+8h
    int row = wm * 32 + mt * 16 + lane16;
    a[mt].u[0] = *(const uint4*)&as[row * 40 + 8 * half];
    a[mt].u[1] = *(const uint4*)&as[row * 40 + 16 + 8 * half];
  }
#pragma unroll
  for (int nt = 0; nt < 2; nt++) {  // B frag: lane=N row, K runs 8h / 16+8h
    int nr = wn * 32 + nt * 16 + lane16;
    b.u[0] = *(const uint4*)&bs[nr * 40 + 8 * half];
    b.u[1] = *(const uint4*)&bs[nr * 40 + 16 + 8 * half];
#pragma unroll
    for (int mt = 0; mt < 2; mt++)
      acc[mt][nt] = wmma_bf16(a[mt], b, acc[mt][nt]);
  }
}

__global__ __launch_bounds__(256) void gemm_bf16_nt(
    const u16* __restrict__ A, const u16* __restrict__ B,
    float* __restrict__ C, int M, int N, int K) {
  __shared__ u16 As[3][64 * 40];   // 64x32, padded stride 40, triple buffered
  __shared__ u16 Bs[3][128 * 40];  // 128x32 (rows are N, cols are K)
  const int tid = threadIdx.x;
  const int wave = tid >> 5, lane = tid & 31;
  const int lane16 = lane & 15, half = lane >> 4;
  const int wm = wave & 1, wn = wave >> 1;  // 2(M) x 4(N) waves
  const int m0 = blockIdx.y * 64;
  const int n0 = blockIdx.x * 128;

  // Per-thread 16B staging slots: A one chunk, B two chunks.
  const int ar = tid >> 2, ac = (tid & 3) * 8;   // A rows 0..63
  const int br0 = tid >> 2;                      // B rows 0..63
  const int br1 = 64 + (tid >> 2);               // B rows 64..127
  const u16* gA  = A + (size_t)(m0 + ar) * K + ac;
  const u16* gB0 = B + (size_t)(n0 + br0) * K + ac;
  const u16* gB1 = B + (size_t)(n0 + br1) * K + ac;
  const int lofsA  = ar * 40 + ac;
  const int lofsB0 = br0 * 40 + ac;
  const int lofsB1 = br1 * 40 + ac;

  v8f acc[2][2] = {};

#if HAVE_ASYNC_LDS
  const int nk = K >> 5;  // K/32 tiles (>= 2 for all our shapes)
  // Prologue: tiles 0 and 1 in flight.
  async_copy16(gA,       &As[0][lofsA]);
  async_copy16(gB0,      &Bs[0][lofsB0]);
  async_copy16(gB1,      &Bs[0][lofsB1]);
  async_copy16(gA + 32,  &As[1][lofsA]);
  async_copy16(gB0 + 32, &Bs[1][lofsB0]);
  async_copy16(gB1 + 32, &Bs[1][lofsB1]);

  int ci = 0;  // compute buffer (tile it)
  int ii = 2;  // issue buffer (tile it+2) == buffer consumed last iteration
  const u16* pA  = gA + 64;
  const u16* pB0 = gB0 + 64;
  const u16* pB1 = gB1 + 64;

  // Steady state: branch-free body, 2 tiles always in flight.
  for (int it = 0; it < nk - 2; ++it) {
    async_copy16(pA,  &As[ii][lofsA]);
    async_copy16(pB0, &Bs[ii][lofsB0]);
    async_copy16(pB1, &Bs[ii][lofsB1]);
    pA += 32; pB0 += 32; pB1 += 32;
    wait_async_le6();  // tile `it` landed; it+1, it+2 still in flight
    __syncthreads();
    mma_step(&As[ci][0], &Bs[ci][0], wm, wn, lane16, half, acc);
    __syncthreads();   // readers done before buffer is refilled
    ci = (ci == 2) ? 0 : ci + 1;
    ii = (ii == 2) ? 0 : ii + 1;
  }
  // Tail: tiles nk-2 and nk-1 (no further issues).
  wait_async_le3();
  __syncthreads();
  mma_step(&As[ci][0], &Bs[ci][0], wm, wn, lane16, half, acc);
  ci = (ci == 2) ? 0 : ci + 1;
  wait_async_0();
  __syncthreads();
  mma_step(&As[ci][0], &Bs[ci][0], wm, wn, lane16, half, acc);
#else
  // Register-mediated double buffer fallback.
  int buf = 0;
  uint4 ra = *(const uint4*)gA;
  uint4 rb0 = *(const uint4*)gB0;
  uint4 rb1 = *(const uint4*)gB1;
  for (int k0 = 0; k0 < K; k0 += 32) {
    *(uint4*)&As[buf][lofsA] = ra;
    *(uint4*)&Bs[buf][lofsB0] = rb0;
    *(uint4*)&Bs[buf][lofsB1] = rb1;
    if (k0 + 32 < K) {
      ra = *(const uint4*)(gA + k0 + 32);
      rb0 = *(const uint4*)(gB0 + k0 + 32);
      rb1 = *(const uint4*)(gB1 + k0 + 32);
      __builtin_prefetch(gA + k0 + 64, 0, 1);
      __builtin_prefetch(gB0 + k0 + 64, 0, 1);
    }
    __syncthreads();
    mma_step(&As[buf][0], &Bs[buf][0], wm, wn, lane16, half, acc);
    __syncthreads();
    buf ^= 1;
  }
#endif

#pragma unroll
  for (int mt = 0; mt < 2; mt++)
#pragma unroll
    for (int nt = 0; nt < 2; nt++) {
      int rowb = m0 + wm * 32 + mt * 16 + 8 * half;
      int colb = n0 + wn * 32 + nt * 16 + lane16;
#pragma unroll
      for (int r = 0; r < 8; r++)
        C[(size_t)(rowb + r) * N + colb] = acc[mt][nt][r];
    }
}

// ------------- postproc: rope + rms on q/k, ve-gate on v -------------------
DEV void rope_rms_store(const float* __restrict__ src, u16* __restrict__ dst,
                        const float* __restrict__ cosb,
                        const float* __restrict__ sinb, int st, int lane) {
  float y[4];
  float ss = 0.f;
#pragma unroll
  for (int i = 0; i < 2; i++) {
    int p = lane + 32 * i;            // rotary pair index (0..63)
    float x1 = src[p], x2 = src[p + 64];
    float c = cosb[(size_t)st * 64 + p], s = sinb[(size_t)st * 64 + p];
    float y1 = x1 * c + x2 * s;
    float y2 = -x1 * s + x2 * c;
    y[i] = y1; y[i + 2] = y2;
    ss += y1 * y1 + y2 * y2;
  }
#pragma unroll
  for (int m = 1; m < 32; m <<= 1) ss += __shfl_xor(ss, m, 32);
  float r = rsqrtf(ss * (1.f / 128.f) + RMSEPS) * QKGAIN;
  dst[lane]      = f2bf(y[0] * r);
  dst[lane + 32] = f2bf(y[1] * r);
  dst[lane + 64] = f2bf(y[2] * r);
  dst[lane + 96] = f2bf(y[3] * r);
}

__global__ __launch_bounds__(256) void postproc_kernel(
    const float* __restrict__ qkv, const float* __restrict__ x,
    const float* __restrict__ ve, const float* __restrict__ cosb,
    const float* __restrict__ sinb, const float* __restrict__ wgate,
    u16* __restrict__ qb, u16* __restrict__ kb, u16* __restrict__ vtb) {
  int wid = (blockIdx.x * blockDim.x + threadIdx.x) >> 5;
  int lane = threadIdx.x & 31;
  const int SLOTS = NH + NKV;  // 24
  int token = wid / SLOTS, slot = wid - token * SLOTS;
  if (token >= MTOK) return;
  int bb = token / SEQ, st = token - bb * SEQ;

  if (slot < NH) {
    int h = slot;
    const float* qr = qkv + (size_t)token * NQKV + h * DH;
    u16* dst = qb + ((size_t)(bb * NH + h) * SEQ + st) * DH;
    rope_rms_store(qr, dst, cosb, sinb, st, lane);
  } else {
    int kv = slot - NH;
    const float* kr = qkv + (size_t)token * NQKV + EMB + kv * DH;
    u16* kdst = kb + ((size_t)(bb * NKV + kv) * SEQ + st) * DH;
    rope_rms_store(kr, kdst, cosb, sinb, st, lane);
    // gate = 3*sigmoid(x[:,:12] . Wgate[kv])
    float g = 0.f;
    if (lane < 12) g = x[(size_t)token * EMB + lane] * wgate[kv * 12 + lane];
#pragma unroll
    for (int m = 1; m < 32; m <<= 1) g += __shfl_xor(g, m, 32);
    g = 3.f / (1.f + __expf(-g));
    // V: add gate*ve, store transposed [b][kv][d][s]
    const float* vr = qkv + (size_t)token * NQKV + EMB + NKV * DH + kv * DH;
    const float* ver = ve + (size_t)token * (NKV * DH) + kv * DH;
#pragma unroll
    for (int i = 0; i < 4; i++) {
      int e = lane + 32 * i;
      float vv = vr[e] + g * ver[e];
      vtb[((size_t)(bb * NKV + kv) * DH + e) * SEQ + st] = f2bf(vv);
    }
  }
}

// -------------------- flash attention, one wave per 16-q tile --------------
__global__ __launch_bounds__(256) void attn_kernel(
    const u16* __restrict__ qb, const u16* __restrict__ kb,
    const u16* __restrict__ vtb, u16* __restrict__ yb,
    const int* __restrict__ wptr) {
  __shared__ u16 pshare[8 * 16 * 32];  // 1KB per wave: P tile 16x32 bf16
  const int wave = threadIdx.x >> 5, lane = threadIdx.x & 31;
  const int lane16 = lane & 15, half = lane >> 4;
  u16* pl = pshare + wave * 512;
  const int window = *wptr;

  const int QB = SEQ / 16;  // 128 q tiles per head
  int gid = blockIdx.x * 8 + wave;
  int bh = gid / QB, qt = gid - bh * QB;
  int bb = bh >> 4, h = bh & 15, kv = h >> 1;  // n_rep = 2
  int q0 = qt * 16;

  const u16* qp = qb + ((size_t)(bb * NH + h) * SEQ + q0) * DH;
  const u16* kp = kb + ((size_t)(bb * NKV + kv) * SEQ) * DH;
  const u16* vp = vtb + ((size_t)(bb * NKV + kv) * DH) * SEQ;

  // Q fragments for full head dim (4 k-steps of 32), kept in registers
  Frag qf[4];
#pragma unroll
  for (int ks = 0; ks < 4; ks++) {
    const u16* base = qp + (size_t)lane16 * DH + ks * 32;
    qf[ks].u[0] = *(const uint4*)(base + 8 * half);
    qf[ks].u[1] = *(const uint4*)(base + 16 + 8 * half);
  }

  float mr[8], lr[8];
  v8f o[8] = {};
#pragma unroll
  for (int r = 0; r < 8; r++) { mr[r] = -1e30f; lr[r] = 0.f; }

  int kstart = q0 - window + 1;
  if (kstart < 0) kstart = 0;
  kstart &= ~31;

  for (int kb0 = kstart; kb0 <= q0 + 15; kb0 += 32) {
    // S = Q . K^T for 32 keys (two 16-key tiles)
    v8f s0 = {}, s1 = {};
#pragma unroll
    for (int ks = 0; ks < 4; ks++) {
      Frag b0, b1;
      const u16* kbase = kp + (size_t)(kb0 + lane16) * DH + ks * 32 + 8 * half;
      b0.u[0] = *(const uint4*)kbase;
      b0.u[1] = *(const uint4*)(kbase + 16);
      const u16* kbase1 = kbase + 16 * DH;
      b1.u[0] = *(const uint4*)kbase1;
      b1.u[1] = *(const uint4*)(kbase1 + 16);
      s0 = wmma_bf16(qf[ks], b0, s0);
      s1 = wmma_bf16(qf[ks], b1, s1);
    }
    // online softmax per row (rows r+8*half; reductions over 16-lane halves)
#pragma unroll
    for (int r = 0; r < 8; r++) {
      int qrow = q0 + r + 8 * half;
      int key0 = kb0 + lane16, key1 = key0 + 16;
      bool ok0 = (key0 <= qrow) && (qrow - key0 < window);
      bool ok1 = (key1 <= qrow) && (qrow - key1 < window);
      float v0 = ok0 ? s0[r] * ATT_SCALE : -1e30f;
      float v1 = ok1 ? s1[r] * ATT_SCALE : -1e30f;
      float mx = fmaxf(v0, v1);
#pragma unroll
      for (int m = 1; m < 16; m <<= 1) mx = fmaxf(mx, __shfl_xor(mx, m, 32));
      float mnew = fmaxf(mr[r], mx);
      float corr = __expf(mr[r] - mnew);
      float p0 = ok0 ? __expf(v0 - mnew) : 0.f;
      float p1 = ok1 ? __expf(v1 - mnew) : 0.f;
      float ps = p0 + p1;
#pragma unroll
      for (int m = 1; m < 16; m <<= 1) ps += __shfl_xor(ps, m, 32);
      lr[r] = lr[r] * corr + ps;
      mr[r] = mnew;
#pragma unroll
      for (int dt = 0; dt < 8; dt++) o[dt][r] *= corr;
      int row = r + 8 * half;
      pl[row * 32 + lane16] = f2bf(p0);
      pl[row * 32 + 16 + lane16] = f2bf(p1);
    }
    // P (16x32) . V (32 keys x 128 dims, V stored transposed [d][s])
    Frag pf;
    {
      const u16* pb = pl + lane16 * 32;
      pf.u[0] = *(const uint4*)(pb + 8 * half);
      pf.u[1] = *(const uint4*)(pb + 16 + 8 * half);
    }
#pragma unroll
    for (int dt = 0; dt < 8; dt++) {
      Frag vf;
      const u16* vb = vp + (size_t)(dt * 16 + lane16) * SEQ + kb0 + 8 * half;
      vf.u[0] = *(const uint4*)vb;
      vf.u[1] = *(const uint4*)(vb + 16);
      o[dt] = wmma_bf16(pf, vf, o[dt]);
    }
  }

  // epilogue: normalize and write y[b][s][h*128+d] as bf16 for proj GEMM
#pragma unroll
  for (int dt = 0; dt < 8; dt++) {
#pragma unroll
    for (int r = 0; r < 8; r++) {
      int tok = q0 + r + 8 * half;
      float val = o[dt][r] / lr[r];
      yb[((size_t)(bb * SEQ + tok)) * EMB + h * DH + dt * 16 + lane16] =
          f2bf(val);
    }
  }
}

// --------------------------------------------------------------------------
extern "C" void kernel_launch(void* const* d_in, const int* in_sizes, int n_in,
                              void* d_out, int out_size, void* d_ws,
                              size_t ws_size, hipStream_t stream) {
  (void)in_sizes; (void)n_in; (void)out_size; (void)ws_size;
  const float* x     = (const float*)d_in[0];
  const float* ve    = (const float*)d_in[1];
  const float* cosb  = (const float*)d_in[2];
  const float* sinb  = (const float*)d_in[3];
  const float* Wq    = (const float*)d_in[4];
  const float* Wk    = (const float*)d_in[5];
  const float* Wv    = (const float*)d_in[6];
  const float* Wproj = (const float*)d_in[7];
  const float* Wgate = (const float*)d_in[8];
  const int*   wsz   = (const int*)d_in[9];

  char* ws = (char*)d_ws;
  u16*   xb    = (u16*)(ws);                   // 4096x2048 bf16  (16 MB)
  u16*   wqkv  = (u16*)(ws + 16777216);        // 4096x2048 bf16  (16 MB)
  u16*   wproj = (u16*)(ws + 33554432);        // 2048x2048 bf16  ( 8 MB)
  float* qkv   = (float*)(ws + 41943040);      // 4096x4096 f32   (64 MB)
  u16*   qbuf  = (u16*)(ws + 109051904);       // [2][16][2048][128] bf16
  u16*   kbuf  = (u16*)(ws + 125829120);       // [2][8][2048][128]  bf16
  u16*   vtbuf = (u16*)(ws + 134217728);       // [2][8][128][2048]  bf16
  u16*   ybuf  = (u16*)(ws + 142606336);       // 4096x2048 bf16  (16 MB)

  auto cast = [&](const float* s, u16* d, int n) {
    cast_bf16_kernel<<<(n + 255) / 256, 256, 0, stream>>>(s, d, n);
  };
  cast(x, xb, MTOK * EMB);
  cast(Wq, wqkv, EMB * EMB);
  cast(Wk, wqkv + (size_t)EMB * EMB, NKV * DH * EMB);
  cast(Wv, wqkv + (size_t)(EMB + NKV * DH) * EMB, NKV * DH * EMB);
  cast(Wproj, wproj, EMB * EMB);

  // QKV projection: [4096,2048] x [4096,2048]^T -> [4096,4096]
  gemm_bf16_nt<<<dim3(NQKV / 128, MTOK / 64), 256, 0, stream>>>(
      xb, wqkv, qkv, MTOK, NQKV, EMB);

  // rope/rms/gate
  postproc_kernel<<<(MTOK * (NH + NKV)) / 8, 256, 0, stream>>>(
      qkv, x, ve, cosb, sinb, Wgate, qbuf, kbuf, vtbuf);

  // flash attention: 2*16*128 waves / 8 waves per block
  attn_kernel<<<(BATCH * NH * (SEQ / 16)) / 8, 256, 0, stream>>>(
      qbuf, kbuf, vtbuf, ybuf, wsz);

  // output projection: [4096,2048] x [2048,2048]^T -> d_out (f32)
  gemm_bf16_nt<<<dim3(EMB / 128, MTOK / 64), 256, 0, stream>>>(
      ybuf, wproj, (float*)d_out, MTOK, EMB, EMB);
}
